// SEEL_3899830305190
// MI455X (gfx1250) — matrix operand order repaired and checked by reading, not compiled, not run on hardware
//
#include <hip/hip_runtime.h>

// ---------------------------------------------------------------------------
// Problem constants (from reference)
// ---------------------------------------------------------------------------
#define N_ROWS   4096
#define DIM      768
#define N_CLASS  10
#define INV_TEMP 10.0f      // 1 / 0.1
#define FIXED_M  10.0f      // |sim| <= 1/temp, safe fixed logsumexp max
#define EPS_NORM 1e-8f

#define KCHUNKS  (DIM / 32)   // 24
#define NJCHUNK  8            // j split: 8 chunks of 512 cols (additive merge)
#define JCHUNK_W (N_ROWS / NJCHUNK)

typedef _Float16 v8h  __attribute__((ext_vector_type(8)));
typedef _Float16 v16h __attribute__((ext_vector_type(16)));
typedef float    v8f  __attribute__((ext_vector_type(8)));
typedef int      v4i  __attribute__((vector_size(16)));   // matches builtin sig

typedef __attribute__((address_space(1))) v4i* gv4i_ptr;
typedef __attribute__((address_space(3))) v4i* lv4i_ptr;

// ---------------------------------------------------------------------------
// CDNA5 async global->LDS copy (16B per lane), ASYNCcnt-tracked
// ---------------------------------------------------------------------------
__device__ __forceinline__ void async_copy16(const _Float16* g, _Float16* l) {
#if __has_builtin(__builtin_amdgcn_global_load_async_to_lds_b128)
  __builtin_amdgcn_global_load_async_to_lds_b128((gv4i_ptr)g, (lv4i_ptr)l, 0, 0);
#else
  unsigned ldsOff = (unsigned)(size_t)(__attribute__((address_space(3))) char*)l;
  asm volatile("global_load_async_to_lds_b128 %0, %1, off"
               :: "v"(ldsOff), "v"(g) : "memory");
#endif
}

__device__ __forceinline__ void wait_async0() {
#if __has_builtin(__builtin_amdgcn_s_wait_asynccnt)
  __builtin_amdgcn_s_wait_asynccnt(0);
#else
  asm volatile("s_wait_asynccnt 0" ::: "memory");
#endif
}

// A-fragment (16x32 f16): per-lane contiguous K (matches B from LDS):
//   lanes 0-15 : elems 0..15 = K kc*32+0..15
//   lanes16-31 : elems 0..15 = K kc*32+16..31
__device__ __forceinline__ v16h load_a_frag(const _Float16* p) {
  v8h lo = *(const v8h*)(p);
  v8h hi = *(const v8h*)(p + 8);
  return __builtin_shufflevector(lo, hi, 0, 1, 2, 3, 4, 5, 6, 7,
                                         8, 9, 10, 11, 12, 13, 14, 15);
}

// ---------------------------------------------------------------------------
// Zero the reduction workspace (3*N per-row partials + 2 scalar accumulators)
// ---------------------------------------------------------------------------
__global__ void scl_init_kernel(float* acc, int n) {
  int i = blockIdx.x * blockDim.x + threadIdx.x;
  if (i < n) acc[i] = 0.0f;
}

// ---------------------------------------------------------------------------
// Row-normalize features -> f16 (one wave per row)
// ---------------------------------------------------------------------------
__global__ __launch_bounds__(256) void norm_kernel(
    const float* __restrict__ feat, _Float16* __restrict__ fn) {
  const int row  = blockIdx.x * 8 + (threadIdx.x >> 5);
  const int lane = threadIdx.x & 31;
  const float* f = feat + (size_t)row * DIM;
  float ss = 0.0f;
  for (int d = lane; d < DIM; d += 32) { float x = f[d]; ss = fmaf(x, x, ss); }
  #pragma unroll
  for (int m = 16; m >= 1; m >>= 1) ss += __shfl_xor(ss, m, 32);
  const float sc = 1.0f / fmaxf(sqrtf(ss), EPS_NORM);
  _Float16* o = fn + (size_t)row * DIM;
  for (int d = lane; d < DIM; d += 32) o[d] = (_Float16)(f[d] * sc);
}

// ---------------------------------------------------------------------------
// logits = features @ W + b  (one wave per row) + CE accumulation
// ---------------------------------------------------------------------------
__global__ __launch_bounds__(256) void logits_ce_kernel(
    const float* __restrict__ feat, const float* __restrict__ W,
    const float* __restrict__ bias, const int* __restrict__ labels,
    float* __restrict__ out_logits, float* __restrict__ ce_sum) {
  const int row  = blockIdx.x * 8 + (threadIdx.x >> 5);
  const int lane = threadIdx.x & 31;
  const float* f = feat + (size_t)row * DIM;
  float acc[N_CLASS];
  #pragma unroll
  for (int c = 0; c < N_CLASS; ++c) acc[c] = 0.0f;
  for (int d = lane; d < DIM; d += 32) {
    const float x = f[d];
    const float* w = W + (size_t)d * N_CLASS;
    #pragma unroll
    for (int c = 0; c < N_CLASS; ++c) acc[c] = fmaf(x, w[c], acc[c]);
  }
  #pragma unroll
  for (int m = 16; m >= 1; m >>= 1) {
    #pragma unroll
    for (int c = 0; c < N_CLASS; ++c) acc[c] += __shfl_xor(acc[c], m, 32);
  }
  float lg[N_CLASS];
  #pragma unroll
  for (int c = 0; c < N_CLASS; ++c) lg[c] = acc[c] + bias[c];
  if (lane < N_CLASS) out_logits[(size_t)row * N_CLASS + lane] = lg[lane];
  if (lane == 0) {
    float mx = lg[0];
    #pragma unroll
    for (int c = 1; c < N_CLASS; ++c) mx = fmaxf(mx, lg[c]);
    float se = 0.0f;
    #pragma unroll
    for (int c = 0; c < N_CLASS; ++c) se += __expf(lg[c] - mx);
    const float lz = mx + __logf(se);
    atomicAdd(ce_sum, lz - lg[labels[row]]);
  }
}

// ---------------------------------------------------------------------------
// Main SCL kernel: fused fn @ fn^T (WMMA) + fixed-max logsumexp + positives.
// One wave handles a 16-row i-strip x 512-col j-chunk; the 4 waves of a block
// share one j-chunk. B K-chunks (64 rows x 32 f16 = 4KB) are staged in LDS
// once per block via async global->LDS DMA, double-buffered: iteration kc
// issues the fill for kc+1, computes from LDS, then s_wait_asynccnt+barrier.
// ---------------------------------------------------------------------------
__global__ __launch_bounds__(128) void scl_main_kernel(
    const _Float16* __restrict__ fn, const int* __restrict__ labels,
    float* __restrict__ lsumArr, float* __restrict__ psumArr,
    float* __restrict__ ncntArr) {
  __shared__ _Float16 sB[2][64 * 32];                  // 2 x 4KB B chunk

  const int lane   = threadIdx.x & 31;
  const int w      = threadIdx.x >> 5;                 // wave in block, 0..3
  const int jchunk = blockIdx.x >> 6;                  // 0..7 (shared by block)
  const int strip  = ((blockIdx.x & 63) << 2) + w;     // 0..255
  const int iBase  = strip * 16;
  const int jBeg   = jchunk * JCHUNK_W;
  const int half   = lane >> 4;
  const int lrow   = lane & 15;

  // cooperative fill indices: 128 threads x 32B = 4KB
  const int fr = (int)threadIdx.x >> 1;                // row 0..63
  const int fh = ((int)threadIdx.x & 1) * 16;          // 16-elem half

  // labels for my 8 C-fragment rows: i = iBase + v + 8*half
  int labI[8];
  #pragma unroll
  for (int v = 0; v < 8; ++v) labI[v] = labels[iBase + v + 8 * half];

  const _Float16* aRow = fn + (size_t)(iBase + lrow) * DIM + half * 16;

  float lsum[8], psum[8], ncnt[8];
  #pragma unroll
  for (int v = 0; v < 8; ++v) { lsum[v] = 0.0f; psum[v] = 0.0f; ncnt[v] = 0.0f; }

  for (int jb = jBeg; jb < jBeg + JCHUNK_W; jb += 64) {
    // warm WGP$ for the next j-block (gfx1250 global_prefetch)
    if (jb + 64 < jBeg + JCHUNK_W) {
      #pragma unroll
      for (int t = 0; t < 4; ++t)
        __builtin_prefetch(fn + (size_t)(jb + 64 + t * 16 + lrow) * DIM, 0, 1);
    }

    int labJ[4];
    v8f acc[4];
    #pragma unroll
    for (int t = 0; t < 4; ++t) {
      labJ[t] = labels[jb + t * 16 + lrow];
      acc[t]  = (v8f){};
    }

    // prologue: async-fill buffer 0 with K-chunk 0
    {
      const _Float16* g = fn + (size_t)(jb + fr) * DIM + fh;
      _Float16* l = &sB[0][fr * 32 + fh];
      async_copy16(g, l);
      async_copy16(g + 8, l + 8);
    }
    wait_async0();
    __syncthreads();

    for (int kc = 0; kc < KCHUNKS; ++kc) {
      const int p = kc & 1;
      // issue async fill for kc+1 into the other buffer (overlaps compute)
      if (kc + 1 < KCHUNKS) {
        const _Float16* g = fn + (size_t)(jb + fr) * DIM + (kc + 1) * 32 + fh;
        _Float16* l = &sB[p ^ 1][fr * 32 + fh];
        async_copy16(g, l);
        async_copy16(g + 8, l + 8);
      }
      // compute from LDS buffer p
      const v16h A = load_a_frag(aRow + kc * 32);
      #pragma unroll
      for (int t = 0; t < 4; ++t) {
        const v16h B = *(const v16h*)&sB[p][(t * 16 + lrow) * 32 + half * 16];
        acc[t] = __builtin_amdgcn_wmma_f32_16x16x32_f16(
            false, A, false, B, (short)0, acc[t], false, false);
      }
      // publish next buffer: own async done, then block-wide barrier
      wait_async0();
      __syncthreads();
    }

    // epilogue: diagonal mask, exp(sim - 10), positive-label accumulation
    #pragma unroll
    for (int t = 0; t < 4; ++t) {
      const int j = jb + t * 16 + lrow;
      #pragma unroll
      for (int v = 0; v < 8; ++v) {
        const int i = iBase + v + 8 * half;
        const float s = acc[t][v] * INV_TEMP;
        if (i != j) {
          lsum[v] += __expf(s - FIXED_M);
          if (labI[v] == labJ[t]) { psum[v] += s; ncnt[v] += 1.0f; }
        }
      }
    }
  }

  // reduce over the 16 lanes of each half (cols of the tile)
  #pragma unroll
  for (int m = 1; m < 16; m <<= 1) {
    #pragma unroll
    for (int v = 0; v < 8; ++v) {
      lsum[v] += __shfl_xor(lsum[v], m, 32);
      psum[v] += __shfl_xor(psum[v], m, 32);
      ncnt[v] += __shfl_xor(ncnt[v], m, 32);
    }
  }
  if (lrow == 0) {
    #pragma unroll
    for (int v = 0; v < 8; ++v) {
      const int i = iBase + v + 8 * half;
      atomicAdd(&lsumArr[i], lsum[v]);
      atomicAdd(&psumArr[i], psum[v]);
      atomicAdd(&ncntArr[i], ncnt[v]);
    }
  }
}

// ---------------------------------------------------------------------------
// Per-row SCL finalization: per_i = (n_i*logZ_i - p_i)/n_i, accumulate sum
// ---------------------------------------------------------------------------
__global__ __launch_bounds__(256) void scl_rows_kernel(
    const float* __restrict__ lsumArr, const float* __restrict__ psumArr,
    const float* __restrict__ ncntArr, float* __restrict__ scl_sum) {
  const int i = blockIdx.x * blockDim.x + threadIdx.x;
  if (i >= N_ROWS) return;
  const float n = ncntArr[i];
  float per = 0.0f;
  if (n > 0.5f) {
    const float logZ = FIXED_M + __logf(lsumArr[i]);
    per = (n * logZ - psumArr[i]) / n;  // = -inner/n
  }
  atomicAdd(scl_sum, per);
}

__global__ void finalize_kernel(const float* __restrict__ sums,
                                float* __restrict__ out) {
  // sums[0] = ce_sum, sums[1] = scl_sum
  out[0] = sums[0] * (1.0f / (float)N_ROWS) +
           0.1f * (sums[1] * (1.0f / (float)N_ROWS));
}

// ---------------------------------------------------------------------------
extern "C" void kernel_launch(void* const* d_in, const int* in_sizes, int n_in,
                              void* d_out, int out_size, void* d_ws,
                              size_t ws_size, hipStream_t stream) {
  const float* features = (const float*)d_in[0];
  const float* W        = (const float*)d_in[1];
  const float* b        = (const float*)d_in[2];
  const int*   labels   = (const int*)d_in[3];
  float* out = (float*)d_out;  // [0] = loss, [1..] = logits (4096 x 10)

  // workspace layout
  _Float16* fn = (_Float16*)d_ws;                               // N*D f16
  float* red = (float*)((char*)d_ws + (size_t)N_ROWS * DIM * 2);
  float* lsumArr = red;                 // [N]
  float* psumArr = red + N_ROWS;        // [N]
  float* ncntArr = red + 2 * N_ROWS;    // [N]
  float* sums    = red + 3 * N_ROWS;    // [2] : ce_sum, scl_sum

  const int nred = 3 * N_ROWS + 2;
  scl_init_kernel<<<(nred + 255) / 256, 256, 0, stream>>>(red, nred);
  norm_kernel<<<N_ROWS / 8, 256, 0, stream>>>(features, fn);
  logits_ce_kernel<<<N_ROWS / 8, 256, 0, stream>>>(features, W, b, labels,
                                                   out + 1, &sums[0]);
  scl_main_kernel<<<(64 * NJCHUNK), 128, 0, stream>>>(fn, labels, lsumArr,
                                                      psumArr, ncntArr);
  scl_rows_kernel<<<N_ROWS / 256, 256, 0, stream>>>(lsumArr, psumArr, ncntArr,
                                                    &sums[1]);
  finalize_kernel<<<1, 1, 0, stream>>>(sums, out);
}
